// NetGCN_40114994545113
// MI455X (gfx1250) — matrix operand order repaired and checked by reading, not compiled, not run on hardware
//
#include <hip/hip_runtime.h>

#define N_NODES 100000
#define N_EDGES 3200000
#define IN_DIM  512   // 64 * 8
#define HID1    256
#define HID2    128
#define ENDD    64

typedef __attribute__((ext_vector_type(16))) __bf16        v16bf;
typedef __attribute__((ext_vector_type(8)))  float         v8f;
typedef __attribute__((ext_vector_type(16))) unsigned short v16u;
typedef __attribute__((ext_vector_type(8)))  unsigned short v8u;

__device__ __forceinline__ unsigned short f2bf(float x) {
  unsigned int u = __float_as_uint(x);
  unsigned int r = u + 0x7FFFu + ((u >> 16) & 1u);   // round-to-nearest-even
  return (unsigned short)(r >> 16);
}

__device__ __forceinline__ v8f wmma_bf16(v16u a, v16u b, v8f c) {
  return __builtin_amdgcn_wmma_f32_16x16x32_bf16(
      false, __builtin_bit_cast(v16bf, a),
      false, __builtin_bit_cast(v16bf, b),
      (short)0, c, false, false);
}

// ---------------- degree / rsqrt ----------------
__global__ void degree_kernel(const int* __restrict__ src, const int* __restrict__ dst,
                              float* __restrict__ degOut, float* __restrict__ degIn, int nE) {
  int i = blockIdx.x * blockDim.x + threadIdx.x;
  if (i < nE) {
    atomicAdd(&degOut[src[i]], 1.0f);
    atomicAdd(&degIn[dst[i]], 1.0f);
  }
}

__global__ void rsqrt_kernel(float* __restrict__ d, int n) {
  int i = blockIdx.x * blockDim.x + threadIdx.x;
  if (i < n) d[i] = rsqrtf(fmaxf(d[i], 1.0f));
}

// ---------------- conversions (fused scaling) ----------------
// Y = bf16(X * rsOut[node])
__global__ void cvt_scale_kernel(const float* __restrict__ X, const float* __restrict__ rs,
                                 unsigned short* __restrict__ Y, int shift, long long total) {
  long long i = (long long)blockIdx.x * blockDim.x + threadIdx.x;
  if (i >= total) return;
  long long node = i >> shift;
  Y[i] = f2bf(X[i] * rs[node]);
}

// Y = bf16( relu(agg*rsIn + b[f]) * rsOut )
__global__ void fin1_kernel(const float* __restrict__ agg, const float* __restrict__ rsIn,
                            const float* __restrict__ rsOut, const float* __restrict__ b,
                            unsigned short* __restrict__ Y, int shift, int mask, long long total) {
  long long i = (long long)blockIdx.x * blockDim.x + threadIdx.x;
  if (i >= total) return;
  long long node = i >> shift;
  int f = (int)(i & mask);
  float v = fmaxf(agg[i] * rsIn[node] + b[f], 0.0f);
  Y[i] = f2bf(v * rsOut[node]);
}

// Y = bf16( agg*rsIn + b[f] )   (no relu, no out-scale: conv2 output feeds head)
__global__ void fin2_kernel(const float* __restrict__ agg, const float* __restrict__ rsIn,
                            const float* __restrict__ b, unsigned short* __restrict__ Y,
                            int shift, int mask, long long total) {
  long long i = (long long)blockIdx.x * blockDim.x + threadIdx.x;
  if (i >= total) return;
  long long node = i >> shift;
  int f = (int)(i & mask);
  Y[i] = f2bf(agg[i] * rsIn[node] + b[f]);
}

// WT[n*K + k] = bf16(W[k*N + n])   (transpose so B-fragment loads are contiguous)
__global__ void wt_cvt_kernel(const float* __restrict__ W, unsigned short* __restrict__ WT,
                              int K, int N) {
  int i = blockIdx.x * blockDim.x + threadIdx.x;
  if (i >= K * N) return;
  int k = i / N, n = i % N;
  WT[(size_t)n * K + k] = f2bf(W[i]);
}

// ---------------- WMMA GEMM: C[M,N] = A[M,K](bf16) @ BT[N,K](bf16)^T ----------------
// One wave per 32x32 C block (2x2 WMMA tiles): A/B fragments each reused twice,
// 4 independent WMMAs per K-step for issue density. M%32==0, N%32==0, K%32==0.
__global__ void gemm_bf16_wmma(const unsigned short* __restrict__ A,
                               const unsigned short* __restrict__ BT,
                               float* __restrict__ C, int M, int N, int K) {
  int wave = threadIdx.x >> 5;
  int lane = threadIdx.x & 31;
  int tn2s = N >> 5;
  int total = (M >> 5) * tn2s;
  int t = blockIdx.x * (blockDim.x >> 5) + wave;
  if (t >= total) return;                  // uniform per wave -> EXEC stays all-1s
  int tm = t / tn2s;
  int tn = t - tm * tn2s;
  int half = lane >> 4;       // 0: lanes 0-15, 1: lanes 16-31
  int l16  = lane & 15;

  const unsigned short* a0p = A + (size_t)(tm * 32 + l16) * K;       // rows  0..15 of block
  const unsigned short* a1p = a0p + (size_t)16 * K;                  // rows 16..31
  const unsigned short* b0p = BT + (size_t)(tn * 32 + l16) * K;      // cols  0..15
  const unsigned short* b1p = b0p + (size_t)16 * K;                  // cols 16..31

  v8f acc00 = {}, acc01 = {}, acc10 = {}, acc11 = {};
  for (int k = 0; k < K; k += 32) {
    // A 16x32 fragment: lane-half selects K block of 16, contiguous in A row (K = half*16 + 0..15)
    v16u a0 = *(const v16u*)(a0p + k + half * 16);
    v16u a1 = *(const v16u*)(a1p + k + half * 16);
    // B 32x16 fragment: lane-half selects K block of 16, contiguous in BT row
    v16u b0 = *(const v16u*)(b0p + k + half * 16);
    v16u b1 = *(const v16u*)(b1p + k + half * 16);
    acc00 = wmma_bf16(a0, b0, acc00);
    acc01 = wmma_bf16(a0, b1, acc01);
    acc10 = wmma_bf16(a1, b0, acc10);
    acc11 = wmma_bf16(a1, b1, acc11);
  }
  // C 16x16 f32 tile: reg r holds M = half*8 + r, N = l16
  float* c00 = C + (size_t)(tm * 32 + half * 8) * N + tn * 32 + l16;
  float* c10 = c00 + (size_t)16 * N;
#pragma unroll
  for (int r = 0; r < 8; ++r) {
    c00[(size_t)r * N]      = acc00[r];
    c00[(size_t)r * N + 16] = acc01[r];
    c10[(size_t)r * N]      = acc10[r];
    c10[(size_t)r * N + 16] = acc11[r];
  }
}

// ---------------- edge scatter: agg[dst] += H[src], F floats per edge ----------------
#define EPB 8
__global__ void scatter_add_kernel(const float* __restrict__ H, const int* __restrict__ src,
                                   const int* __restrict__ dst, float* __restrict__ agg,
                                   int nE) {
  __shared__ int s_src[EPB], s_dst[EPB];
  int F = blockDim.x;
  int e0 = blockIdx.x * EPB;
  if (threadIdx.x < EPB) {
    int e = e0 + threadIdx.x;
    if (e < nE) { s_src[threadIdx.x] = src[e]; s_dst[threadIdx.x] = dst[e]; }
  }
  __syncthreads();
  int e1 = e0 + EPB; if (e1 > nE) e1 = nE;
  for (int e = e0; e < e1; ++e) {
    int s = s_src[e - e0], d = s_dst[e - e0];
    float v = H[(size_t)s * F + threadIdx.x];
    atomicAdd(&agg[(size_t)d * F + threadIdx.x], v);
  }
}

// ---------------- head: out = relu(H3 + bo1) . Wo2 + bo2, one wave per node ----------------
__global__ void head_kernel(const float* __restrict__ H3, const float* __restrict__ bo1,
                            const float* __restrict__ Wo2, const float* __restrict__ bo2,
                            float* __restrict__ out, int n) {
  int wave = threadIdx.x >> 5, lane = threadIdx.x & 31;
  int node = blockIdx.x * (blockDim.x >> 5) + wave;
  if (node >= n) return;
  const float* h = H3 + (size_t)node * ENDD;
  float t = 0.0f;
#pragma unroll
  for (int j = lane; j < ENDD; j += 32) {
    float v = fmaxf(h[j] + bo1[j], 0.0f);
    t += v * Wo2[j];
  }
#pragma unroll
  for (int off = 16; off > 0; off >>= 1) t += __shfl_down(t, off, 32);
  if (lane == 0) out[node] = t + bo2[0];
}

extern "C" void kernel_launch(void* const* d_in, const int* in_sizes, int n_in,
                              void* d_out, int out_size, void* d_ws, size_t ws_size,
                              hipStream_t stream) {
  const float* features = (const float*)d_in[0];
  const int*   src      = (const int*)d_in[1];
  const int*   dst      = (const int*)d_in[2];
  const float* W1  = (const float*)d_in[3];
  const float* b1  = (const float*)d_in[4];
  const float* W2  = (const float*)d_in[5];
  const float* b2  = (const float*)d_in[6];
  const float* Wo1 = (const float*)d_in[7];
  const float* bo1 = (const float*)d_in[8];
  const float* Wo2 = (const float*)d_in[9];
  const float* bo2 = (const float*)d_in[10];
  float* out = (float*)d_out;

  char* ws = (char*)d_ws;
  const size_t MB = 1ull << 20;
  float* rs_out = (float*)(ws);                 // 400 KB
  float* rs_in  = (float*)(ws + 512 * 1024);    // 400 KB
  unsigned short* wT = (unsigned short*)(ws + 1 * MB);  // up to 256 KB
  const size_t OFF_A = 2 * MB;
  const size_t BUF   = 100 * MB;                // >= 100000*512*2 = 100000*256*4
  const size_t OFF_B = OFF_A + BUF;
  char* bufA = ws + OFF_A;
  char* bufB = ws + OFF_B;

  // ---- degrees -> rsqrt(clip(deg,1)) ----
  hipMemsetAsync(rs_out, 0, N_NODES * sizeof(float), stream);
  hipMemsetAsync(rs_in,  0, N_NODES * sizeof(float), stream);
  degree_kernel<<<(N_EDGES + 255) / 256, 256, 0, stream>>>(src, dst, rs_out, rs_in, N_EDGES);
  rsqrt_kernel<<<(N_NODES + 255) / 256, 256, 0, stream>>>(rs_out, N_NODES);
  rsqrt_kernel<<<(N_NODES + 255) / 256, 256, 0, stream>>>(rs_in, N_NODES);

  // ---- layer 1: Xb1 = bf16(features * rs_out)  [bufA] ----
  {
    long long total = (long long)N_NODES * IN_DIM;
    cvt_scale_kernel<<<(int)((total + 255) / 256), 256, 0, stream>>>(
        features, rs_out, (unsigned short*)bufA, 9, total);
  }
  wt_cvt_kernel<<<(IN_DIM * HID1 + 255) / 256, 256, 0, stream>>>(W1, wT, IN_DIM, HID1);
  {
    int tiles = (N_NODES / 32) * (HID1 / 32);
    gemm_bf16_wmma<<<(tiles + 7) / 8, 256, 0, stream>>>(
        (const unsigned short*)bufA, wT, (float*)bufB, N_NODES, HID1, IN_DIM);
  }
  // agg1 in bufA (Xb1 dead)
  hipMemsetAsync(bufA, 0, (size_t)N_NODES * HID1 * sizeof(float), stream);
  scatter_add_kernel<<<(N_EDGES + EPB - 1) / EPB, HID1, 0, stream>>>(
      (const float*)bufB, src, dst, (float*)bufA, N_EDGES);
  // Xb2 = bf16(relu(agg1*rs_in + b1) * rs_out)  [bufB]
  {
    long long total = (long long)N_NODES * HID1;
    fin1_kernel<<<(int)((total + 255) / 256), 256, 0, stream>>>(
        (const float*)bufA, rs_in, rs_out, b1, (unsigned short*)bufB, 8, HID1 - 1, total);
  }

  // ---- layer 2 ----
  wt_cvt_kernel<<<(HID1 * HID2 + 255) / 256, 256, 0, stream>>>(W2, wT, HID1, HID2);
  {
    int tiles = (N_NODES / 32) * (HID2 / 32);
    gemm_bf16_wmma<<<(tiles + 7) / 8, 256, 0, stream>>>(
        (const unsigned short*)bufB, wT, (float*)bufA, N_NODES, HID2, HID1);
  }
  hipMemsetAsync(bufB, 0, (size_t)N_NODES * HID2 * sizeof(float), stream);
  scatter_add_kernel<<<(N_EDGES + EPB - 1) / EPB, HID2, 0, stream>>>(
      (const float*)bufA, src, dst, (float*)bufB, N_EDGES);
  // Xb3 = bf16(agg2*rs_in + b2)  [bufA]
  {
    long long total = (long long)N_NODES * HID2;
    fin2_kernel<<<(int)((total + 255) / 256), 256, 0, stream>>>(
        (const float*)bufB, rs_in, b2, (unsigned short*)bufA, 7, HID2 - 1, total);
  }

  // ---- head GEMM: H3 = Xb3 @ Wo1  [bufB] ----
  wt_cvt_kernel<<<(HID2 * ENDD + 255) / 256, 256, 0, stream>>>(Wo1, wT, HID2, ENDD);
  {
    int tiles = (N_NODES / 32) * (ENDD / 32);
    gemm_bf16_wmma<<<(tiles + 7) / 8, 256, 0, stream>>>(
        (const unsigned short*)bufA, wT, (float*)bufB, N_NODES, ENDD, HID2);
  }
  head_kernel<<<(N_NODES + 7) / 8, 256, 0, stream>>>(
      (const float*)bufB, bo1, Wo2, bo2, out, N_NODES);
}
